// BinaryLSTMCell_44607530336603
// MI455X (gfx1250) — matrix operand order repaired and checked by reading, not compiled, not run on hardware
//
#include <hip/hip_runtime.h>

// ---------------------------------------------------------------------------
// BinaryLSTMCell for MI455X (gfx1250, wave32, WMMA).
//   x_all = hard_tanh(inputs @ sign(K));  r_all = hard_tanh(h) @ sign(RK)
//   gates -> h_new, c_new.   Outputs: [h_new | h_new | c_new]
// Two bf16 WMMA GEMMs (65536x256x1024 each) with fused epilogue.
// ---------------------------------------------------------------------------

typedef __attribute__((ext_vector_type(16))) __bf16 v16bf;
typedef __attribute__((ext_vector_type(8)))  __bf16 v8bf;
typedef __attribute__((ext_vector_type(8)))  float  v8f;

#define UNITS 256
#define DIN   256
#define NCOL  1024        // 4 * UNITS
#define MTILE 128         // rows per workgroup (8 waves x 16 rows)
#define BT_K  264         // 256 K + 8 element pad (row stride 528B: bank 33n mod 64)

__device__ __forceinline__ float clip1(float x) {
  return __builtin_fminf(__builtin_fmaxf(x, -1.0f), 1.0f);
}

// Load one 16x32 bf16 A-fragment for this lane from a row-major f32 row.
// ISA A layout (16-bit, 16x32): lanes 0-15 hold K = [kA..kA+8) and [kA+16..kA+24)
// with kA = 32*s + 8*khalf (khalf = lane>=16).
template <bool CLIP>
__device__ __forceinline__ v16bf load_a_frag(const float* __restrict__ row, int kA) {
  const float4* p0 = (const float4*)(row + kA);
  const float4* p1 = (const float4*)(row + kA + 16);
  float4 c0 = p0[0], c1 = p0[1], c2 = p1[0], c3 = p1[1];
  float v[16] = {c0.x, c0.y, c0.z, c0.w, c1.x, c1.y, c1.z, c1.w,
                 c2.x, c2.y, c2.z, c2.w, c3.x, c3.y, c3.z, c3.w};
  v16bf r;
#pragma unroll
  for (int q = 0; q < 16; ++q) {
    float f = CLIP ? clip1(v[q]) : v[q];
    r[q] = (__bf16)f;
  }
  return r;
}

extern "C" __global__ void __launch_bounds__(256)
blstm_wmma_kernel(const float* __restrict__ xin, const float* __restrict__ hin,
                  const float* __restrict__ cin, const float* __restrict__ wk,
                  const float* __restrict__ wr, float* __restrict__ out, int bsz)
{
  // Binarized, transposed weight panels: [panel][N][K], bf16 +-1.
  // panels 0-3: kernel gates i,f,c,o ; panels 4-7: recurrent gates i,f,c,o
  __shared__ __bf16 Bt[8][16][BT_K];

  const int tid   = threadIdx.x;
  const int wave  = tid >> 5;
  const int lane  = tid & 31;
  const int mlane = lane & 15;
  const int khalf = lane >> 4;

  const int wrow = blockIdx.x * MTILE + wave * 16;   // this wave's 16 rows
  const float* arow = xin + (size_t)(wrow + mlane) * DIN;
  const float* hrow = hin + (size_t)(wrow + mlane) * UNITS;
  const size_t BU = (size_t)bsz * UNITS;

  for (int ut = 0; ut < 16; ++ut) {             // 16 unit-tiles of 16 units
    const int u0 = ut * 16;

    __syncthreads();                            // prior tile's Bt reads done
    // ---- stage 8 weight panels: binarize f32 -> bf16 +-1, transpose to [N][K]
    for (int it = tid; it < 8 * 16 * 32; it += 256) {
      const int n  = it & 15;
      const int k8 = ((it >> 4) & 31) * 8;
      const int p  = it >> 9;
      const float* src  = (p < 4) ? wk : wr;
      const float* colp = src + (size_t)(p & 3) * UNITS + u0 + n;
      v8bf pk;
#pragma unroll
      for (int q = 0; q < 8; ++q) {
        float w = colp[(size_t)(k8 + q) * NCOL];
        pk[q] = (__bf16)((w >= 0.0f) ? 1.0f : -1.0f);
      }
      *(v8bf*)&Bt[p][n][k8] = pk;               // one 16B ds_store per 8 elems
    }
    __syncthreads();

    // ---- accumulate 8 panels over K=256 (8 WMMA steps each)
    v8f acc[8];
#pragma unroll
    for (int p = 0; p < 8; ++p)
#pragma unroll
      for (int j = 0; j < 8; ++j) acc[p][j] = 0.0f;

#pragma unroll
    for (int s = 0; s < 8; ++s) {
      const int kA = s * 32 + khalf * 8;        // A frag (interleaved layout)
      const int kB = s * 32 + khalf * 8;        // B frag (interleaved layout, same as A)
      v16bf af = load_a_frag<false>(arow, kA);  // inputs
      v16bf hf = load_a_frag<true >(hrow, kA);  // hard_tanh(h)
#pragma unroll
      for (int p = 0; p < 4; ++p) {
        v8bf blo = *(const v8bf*)&Bt[p][mlane][kB];
        v8bf bhi = *(const v8bf*)&Bt[p][mlane][kB + 16];
        v16bf bfrag = __builtin_shufflevector(blo, bhi,
                        0,1,2,3,4,5,6,7,8,9,10,11,12,13,14,15);
        acc[p] = __builtin_amdgcn_wmma_f32_16x16x32_bf16(
                     false, af, false, bfrag, (short)0, acc[p], false, false);
      }
#pragma unroll
      for (int p = 4; p < 8; ++p) {
        v8bf blo = *(const v8bf*)&Bt[p][mlane][kB];
        v8bf bhi = *(const v8bf*)&Bt[p][mlane][kB + 16];
        v16bf bfrag = __builtin_shufflevector(blo, bhi,
                        0,1,2,3,4,5,6,7,8,9,10,11,12,13,14,15);
        acc[p] = __builtin_amdgcn_wmma_f32_16x16x32_bf16(
                     false, hf, false, bfrag, (short)0, acc[p], false, false);
      }
    }

    // ---- fused LSTM epilogue. C/D layout: VGPR j -> M = j + 8*khalf, N = mlane
    #pragma unroll
    for (int j = 0; j < 8; ++j) {
      const int row = wrow + j + khalf * 8;
      const int col = u0 + mlane;
      const size_t idx = (size_t)row * UNITS + col;
      float xi = clip1(acc[0][j]);              // x_all is clipped per element
      float xf = clip1(acc[1][j]);
      float xc = clip1(acc[2][j]);
      float xo = clip1(acc[3][j]);
      float ri = acc[4][j], rf = acc[5][j], rc = acc[6][j], ro = acc[7][j];
      float fg = clip1(xi + rf);                // f = ht(x_i + r_f)  (crossed!)
      float ig = clip1(xf + ri);                // i = ht(x_f + r_i)
      float ct = clip1(cin[idx]);               // c_tm1 = ht(c)
      float cn = fg * ct + ig * clip1(xc + rc);
      float og = clip1(xo + ro);
      float hn = clip1(og * clip1(cn));
      out[idx]            = hn;                 // h_new
      out[BU + idx]       = hn;                 // h_new (duplicate output)
      out[2 * BU + idx]   = cn;                 // c_new
    }
  }
}

extern "C" void kernel_launch(void* const* d_in, const int* in_sizes, int n_in,
                              void* d_out, int out_size, void* d_ws, size_t ws_size,
                              hipStream_t stream) {
  const float* xin = (const float*)d_in[0];   // inputs  (B, 256)
  const float* hin = (const float*)d_in[1];   // h       (B, 256)
  const float* cin = (const float*)d_in[2];   // c       (B, 256)
  const float* wk  = (const float*)d_in[3];   // kernel  (256, 1024)
  const float* wr  = (const float*)d_in[4];   // rec.k.  (256, 1024)
  float* out = (float*)d_out;                 // [h_new | h_new | c_new]

  const int bsz  = in_sizes[0] / DIN;         // 65536
  const int grid = (bsz + MTILE - 1) / MTILE; // 512 workgroups
  blstm_wmma_kernel<<<dim3(grid), dim3(256), 0, stream>>>(
      xin, hin, cin, wk, wr, out, bsz);
}
